// DecoderBlock_75462575391207
// MI455X (gfx1250) — compile-verified, matrix operand order checked by reading, not run on hardware
//
#include <hip/hip_runtime.h>

// ---------------- problem constants ----------------
#define Bb   32
#define Tt   512
#define Dd   512
#define Hh   8
#define DH   64
#define FH   2048
#define BT   (Bb * Tt)          // 16384
#define NTD  ((size_t)BT * Dd)  // 8,388,608
#define NTF  ((size_t)BT * FH)  // 33,554,432

typedef __attribute__((ext_vector_type(16))) _Float16 v16h;
typedef __attribute__((ext_vector_type(8)))  _Float16 v8h;
typedef __attribute__((ext_vector_type(4)))  _Float16 v4h;
typedef __attribute__((ext_vector_type(8)))  float    v8f;
typedef __attribute__((ext_vector_type(4)))  float    v4f;

struct h16pair { v8h lo, hi; };

// Load a 16x32 f16 A-fragment (or 32x16 B-fragment of x@W.T, identical per-lane
// pattern): lane<16 holds K {0..7,16..23} of row (lane&15); lane>=16 holds
// K {8..15,24..31}. Two contiguous 16B loads per lane.
__device__ __forceinline__ v16h frag_ld(const _Float16* p0, int ld) {
    const int lane = threadIdx.x & 31;
    const int m  = lane & 15;
    const int ko = (lane >> 4) << 3;           // 0 or 8
    const _Float16* p = p0 + (size_t)m * ld + ko;
    h16pair t;
    t.lo = *(const v8h*)(p);
    t.hi = *(const v8h*)(p + 16);
    return __builtin_bit_cast(v16h, t);
}

__device__ __forceinline__ v8f wmma16(v16h a, v16h b, v8f c) {
    return __builtin_amdgcn_wmma_f32_16x16x32_f16(false, a, false, b,
                                                  (short)0, c, false, false);
}

// ---------------- fp32 -> f16 convert (vectorized x4) ----------------
__global__ void k_cvt4(const float* __restrict__ s, _Float16* __restrict__ d, int n4) {
    int i = blockIdx.x * 256 + threadIdx.x;
    if (i < n4) {
        v4f v = *(const v4f*)(s + (size_t)i * 4);
        v4h o;
        o[0] = (_Float16)v[0]; o[1] = (_Float16)v[1];
        o[2] = (_Float16)v[2]; o[3] = (_Float16)v[3];
        *(v4h*)(d + (size_t)i * 4) = o;
    }
}

// ---------------- generic GEMM: C[M,N] = A[M,K] @ W[N,K]^T ----------------
// one wave computes a 32x64 tile (2 M-tiles x 4 N-tiles); ping-pong pipelined
// (no register rotation). K/32 must be even (512 / 2048 here).
// mode: 0 = plain f16 store, 1 = +bias & LeakyReLU(0.01) f16 store,
// 2 = f16 store transposed per head into Vt[b,h,d,t] (V projection).
__global__ void __launch_bounds__(32, 1)
k_gemm(const _Float16* __restrict__ A,
       const _Float16* __restrict__ W,
       _Float16* __restrict__ O,
       int N, int K, const float* __restrict__ bias, int mode) {
    const int m0 = blockIdx.x * 32;
    const int n0 = blockIdx.y * 64;
    const int lane = threadIdx.x & 31;
    const int col16 = lane & 15, hi = lane >> 4;

    v8f acc[2][4];
#pragma unroll
    for (int i = 0; i < 2; ++i)
#pragma unroll
        for (int j = 0; j < 4; ++j)
#pragma unroll
            for (int e = 0; e < 8; ++e) acc[i][j][e] = 0.0f;

    const _Float16* Arow0 = A + (size_t)m0 * K;
    const _Float16* Arow1 = A + (size_t)(m0 + 16) * K;
    const _Float16* Wn    = W + (size_t)n0 * K;

    v16h A0[2], A1[2], B0[4], B1[4];

    auto loadAB = [&](v16h* a, v16h* b, int k0) {
        a[0] = frag_ld(Arow0 + k0, K);
        a[1] = frag_ld(Arow1 + k0, K);
#pragma unroll
        for (int nt = 0; nt < 4; ++nt)
            b[nt] = frag_ld(Wn + (size_t)(nt * 16) * K + k0, K);
    };
    auto mm = [&](const v16h* a, const v16h* b) {
#pragma unroll
        for (int mt = 0; mt < 2; ++mt)
#pragma unroll
            for (int nt = 0; nt < 4; ++nt)
                acc[mt][nt] = wmma16(a[mt], b[nt], acc[mt][nt]);
    };

    loadAB(A0, B0, 0);
    for (int k = 0; k + 128 <= K; k += 64) {
        loadAB(A1, B1, k + 32);
        mm(A0, B0);
        loadAB(A0, B0, k + 64);
        mm(A1, B1);
    }
    loadAB(A1, B1, K - 32);   // tail pair: k = K-64 already in A0/B0
    mm(A0, B0);
    mm(A1, B1);

#pragma unroll
    for (int mt = 0; mt < 2; ++mt)
#pragma unroll
        for (int nt = 0; nt < 4; ++nt) {
            const int n = n0 + nt * 16 + col16;
            float bv = (mode == 1) ? bias[n] : 0.0f;
#pragma unroll
            for (int r = 0; r < 8; ++r) {
                const int m = m0 + mt * 16 + r + 8 * hi;
                float v = acc[mt][nt][r];
                if (mode == 1) { v += bv; v = (v > 0.0f) ? v : 0.01f * v; }
                if (mode == 2) {
                    int bbx = m / Tt, t = m - bbx * Tt;
                    int h = n >> 6, d = n & 63;
                    O[(((size_t)bbx * Hh + h) * DH + d) * Tt + t] = (_Float16)v;
                } else {
                    O[(size_t)m * N + n] = (_Float16)v;
                }
            }
        }
}

// ---------------- fused GEMM + residual + LayerNorm ----------------
// block = 128 threads (4 waves); block owns 16 full rows of width D=512.
// Yout = LN(R + A@W^T (+bias)) ; writes fp32 and optional f16.
__global__ void __launch_bounds__(128, 1)
k_gemm_ln(const _Float16* __restrict__ A,
          const _Float16* __restrict__ W, int K,
          const float* __restrict__ bias,
          const float* __restrict__ R,
          const float* __restrict__ gamma,
          const float* __restrict__ beta,
          float* __restrict__ Y32,
          _Float16* __restrict__ Y16) {
    __shared__ __align__(16) float sRow[16][Dd + 4];
    __shared__ float sP1[16][8], sP2[16][8];
    __shared__ float sMean[16], sRstd[16];

    const int tid = threadIdx.x;
    const int wave = tid >> 5, lane = tid & 31;
    const int col16 = lane & 15, hi = lane >> 4;
    const int m0 = blockIdx.x * 16;
    const int n0 = wave * 128;

    v8f acc[8];
#pragma unroll
    for (int i = 0; i < 8; ++i)
#pragma unroll
        for (int j = 0; j < 8; ++j) acc[i][j] = 0.0f;

    const _Float16* Arow = A + (size_t)m0 * K;
    const _Float16* Wn   = W + (size_t)n0 * K;

    v16h a0, a1, B0[8], B1[8];

    auto loadAB = [&](v16h& a, v16h* b, int k0) {
        a = frag_ld(Arow + k0, K);
#pragma unroll
        for (int nt = 0; nt < 8; ++nt)
            b[nt] = frag_ld(Wn + (size_t)(nt * 16) * K + k0, K);
    };
    auto mm = [&](const v16h& a, const v16h* b) {
#pragma unroll
        for (int nt = 0; nt < 8; ++nt)
            acc[nt] = wmma16(a, b[nt], acc[nt]);
    };

    loadAB(a0, B0, 0);
    for (int k = 0; k + 128 <= K; k += 64) {
        loadAB(a1, B1, k + 32);
        mm(a0, B0);
        loadAB(a0, B0, k + 64);
        mm(a1, B1);
    }
    loadAB(a1, B1, K - 32);
    mm(a0, B0);
    mm(a1, B1);

#pragma unroll
    for (int nt = 0; nt < 8; ++nt) {
        const int n = n0 + nt * 16 + col16;
        const float bv = bias ? bias[n] : 0.0f;
#pragma unroll
        for (int r = 0; r < 8; ++r)
            sRow[r + 8 * hi][n] = acc[nt][r] + bv;
    }
    __syncthreads();

    // residual add + row stats: 8 threads per row, 64 cols each
    const int row = tid & 15;
    const int seg = tid >> 4;
    float s1 = 0.0f, s2 = 0.0f;
    for (int c = seg * 64; c < seg * 64 + 64; ++c) {
        float v = sRow[row][c] + R[(size_t)(m0 + row) * Dd + c];
        sRow[row][c] = v;
        s1 += v; s2 += v * v;
    }
    sP1[row][seg] = s1; sP2[row][seg] = s2;
    __syncthreads();
    if (tid < 16) {
        float a1s = 0.0f, a2s = 0.0f;
#pragma unroll
        for (int s = 0; s < 8; ++s) { a1s += sP1[tid][s]; a2s += sP2[tid][s]; }
        float mean = a1s / (float)Dd;
        float var  = a2s / (float)Dd - mean * mean;
        sMean[tid] = mean;
        sRstd[tid] = rsqrtf(var + 1e-5f);
    }
    __syncthreads();
    const float mean = sMean[row], rstd = sRstd[row];
    for (int c = seg * 64; c < seg * 64 + 64; ++c) {
        float v = (sRow[row][c] - mean) * rstd * gamma[c] + beta[c];
        Y32[(size_t)(m0 + row) * Dd + c] = v;
        if (Y16) Y16[(size_t)(m0 + row) * Dd + c] = (_Float16)v;
    }
}

// ---------------- attention: one wave per (b, h, 16-query tile) ----------------
// scores = (Q Kt)/8 masked -> unnormalized exp (f16, LDS) -> O = (E @ V)/rowsum
// vlMode 0: valid = vl[b*T+q] (self, [B,T]); 1: valid = vl[b] (cross, [B]).
__global__ void __launch_bounds__(32, 1)
k_attn(const _Float16* __restrict__ Q,
       const _Float16* __restrict__ Km,
       const _Float16* __restrict__ Vt,
       _Float16* __restrict__ O,
       const int* __restrict__ vl, int vlMode) {
    __shared__ __align__(16) _Float16 sE[16][Tt + 8];
    __shared__ float sMax[16], sInv[16];
    __shared__ int   sVL[16];

    const int q0 = blockIdx.x * 16;
    const int h  = blockIdx.y;
    const int b  = blockIdx.z;
    const int lane = threadIdx.x & 31;
    const int col16 = lane & 15, hi = lane >> 4;

    if (lane < 16)
        sVL[lane] = (vlMode == 0) ? vl[b * Tt + q0 + lane] : vl[b];
    __syncthreads();

    const _Float16* Qbh = Q  + ((size_t)(b * Tt + q0)) * Dd + h * DH;
    const _Float16* Kbh = Km + ((size_t)b * Tt) * Dd + h * DH;
    const _Float16* Vbh = Vt + ((size_t)(b * Hh + h)) * DH * Tt;

    // ---- phase 1: scores, mask/scale, store f16 to LDS, track row max ----
    v16h a0 = frag_ld(Qbh + 0,  Dd);
    v16h a1 = frag_ld(Qbh + 32, Dd);

    float mx[8];
#pragma unroll
    for (int r = 0; r < 8; ++r) mx[r] = -3.0e38f;

    int vlr[8];
#pragma unroll
    for (int r = 0; r < 8; ++r) vlr[r] = sVL[r + 8 * hi];

    v16h KB0[2], KB1[2];
    auto loadK = [&](v16h* d, int nt) {
        d[0] = frag_ld(Kbh + (size_t)(nt * 16) * Dd + 0,  Dd);
        d[1] = frag_ld(Kbh + (size_t)(nt * 16) * Dd + 32, Dd);
    };
    auto scoreTile = [&](const v16h* kb, int nt) {
        v8f c;
#pragma unroll
        for (int j = 0; j < 8; ++j) c[j] = 0.0f;
        c = wmma16(a0, kb[0], c);
        c = wmma16(a1, kb[1], c);
        const int colk = nt * 16 + col16;
#pragma unroll
        for (int r = 0; r < 8; ++r) {
            float s = c[r] * 0.125f;                 // 1/sqrt(64)
            s = (colk < vlr[r]) ? s : -60000.0f;     // mask
            mx[r] = fmaxf(mx[r], s);
            sE[r + 8 * hi][colk] = (_Float16)s;
        }
    };

    const int NT = Tt / 16;                          // 32 (even)
    loadK(KB0, 0);
    for (int nt = 0; nt + 4 <= NT; nt += 2) {
        loadK(KB1, nt + 1);
        scoreTile(KB0, nt);
        loadK(KB0, nt + 2);
        scoreTile(KB1, nt + 1);
    }
    loadK(KB1, NT - 1);                              // KB0 holds NT-2
    scoreTile(KB0, NT - 2);
    scoreTile(KB1, NT - 1);

    // ---- phase 2: row max across the 16-lane half-groups ----
#pragma unroll
    for (int r = 0; r < 8; ++r) {
#pragma unroll
        for (int off = 1; off < 16; off <<= 1)
            mx[r] = fmaxf(mx[r], __shfl_xor(mx[r], off, 32));
    }
    if (col16 == 0) {
#pragma unroll
        for (int r = 0; r < 8; ++r) sMax[r + 8 * hi] = mx[r];
    }
    __syncthreads();

    // ---- phase 3: exp in place (unnormalized), row sums ----
    {
        const int row = col16;
        const float m = sMax[row];
        float s = 0.0f;
        for (int c = hi * (Tt / 2); c < (hi + 1) * (Tt / 2); ++c) {
            float v = (float)sE[row][c];
            float e = __expf(v - m);
            s += e;
            sE[row][c] = (_Float16)e;
        }
        s += __shfl_xor(s, 16, 32);
        if (hi == 0) sInv[row] = 1.0f / s;
    }
    __syncthreads();

    // ---- phase 4: O = E @ V  (contraction over Tt), ping-pong on V ----
    v8f acc[4];
#pragma unroll
    for (int i = 0; i < 4; ++i)
#pragma unroll
        for (int j = 0; j < 8; ++j) acc[i][j] = 0.0f;

    const _Float16* sEf = &sE[0][0];
    v16h V0[4], V1[4];
    auto loadV = [&](v16h* d, int kt) {
#pragma unroll
        for (int nt = 0; nt < 4; ++nt)
            d[nt] = frag_ld(Vbh + (size_t)(nt * 16) * Tt + kt * 32, Tt);
    };
    auto mmPV = [&](const v16h* vb, int kt) {
        v16h a = frag_ld(sEf + kt * 32, Tt + 8);
#pragma unroll
        for (int nt = 0; nt < 4; ++nt)
            acc[nt] = wmma16(a, vb[nt], acc[nt]);
    };

    const int KT = Tt / 32;                          // 16 (even)
    loadV(V0, 0);
    for (int kt = 0; kt + 4 <= KT; kt += 2) {
        loadV(V1, kt + 1);
        mmPV(V0, kt);
        loadV(V0, kt + 2);
        mmPV(V1, kt + 1);
    }
    loadV(V1, KT - 1);                               // V0 holds KT-2
    mmPV(V0, KT - 2);
    mmPV(V1, KT - 1);

    float inv[8];
#pragma unroll
    for (int r = 0; r < 8; ++r) inv[r] = sInv[r + 8 * hi];

#pragma unroll
    for (int nt = 0; nt < 4; ++nt) {
#pragma unroll
        for (int r = 0; r < 8; ++r) {
            const int q = q0 + r + 8 * hi;
            const int d = nt * 16 + col16;
            O[((size_t)(b * Tt + q)) * Dd + h * DH + d] =
                (_Float16)(acc[nt][r] * inv[r]);
        }
    }
}

// ---------------- host-side orchestration ----------------
extern "C" void kernel_launch(void* const* d_in, const int* in_sizes, int n_in,
                              void* d_out, int out_size, void* d_ws, size_t ws_size,
                              hipStream_t stream) {
    const float* X   = (const float*)d_in[0];
    const float* ENC = (const float*)d_in[1];
    const int*   dvl = (const int*)d_in[2];
    const int*   evl = (const int*)d_in[3];
    const float* Wf[8] = { (const float*)d_in[4],  (const float*)d_in[5],
                           (const float*)d_in[6],  (const float*)d_in[7],
                           (const float*)d_in[8],  (const float*)d_in[9],
                           (const float*)d_in[10], (const float*)d_in[11] };
    const float* W1f = (const float*)d_in[12];
    const float* b1  = (const float*)d_in[13];
    const float* W2f = (const float*)d_in[14];
    const float* b2  = (const float*)d_in[15];
    const float* g1 = (const float*)d_in[16]; const float* be1 = (const float*)d_in[17];
    const float* g2 = (const float*)d_in[18]; const float* be2 = (const float*)d_in[19];
    const float* g3 = (const float*)d_in[20]; const float* be3 = (const float*)d_in[21];
    float* OUT = (float*)d_out;

    // workspace carve-up
    char* ws = (char*)d_ws;
    size_t off = 0;
    auto take = [&](size_t bytes) { char* p = ws + off;
                                    off = (off + bytes + 255) & ~(size_t)255; return p; };
    _Float16* X16 = (_Float16*)take(NTD * 2);
    _Float16* E16 = (_Float16*)take(NTD * 2);
    _Float16* Wh[8];
    for (int i = 0; i < 8; ++i) Wh[i] = (_Float16*)take((size_t)Dd * Dd * 2);
    _Float16* W1h = (_Float16*)take((size_t)FH * Dd * 2);
    _Float16* W2h = (_Float16*)take((size_t)Dd * FH * 2);
    _Float16* Qh  = (_Float16*)take(NTD * 2);
    _Float16* Kh  = (_Float16*)take(NTD * 2);
    _Float16* Vth = (_Float16*)take(NTD * 2);
    _Float16* Oh  = (_Float16*)take(NTD * 2);
    _Float16* Y16 = (_Float16*)take(NTD * 2);   // also reused as Z16
    float*    Y32 = (float*)   take(NTD * 4);   // also reused (in place) as Z32
    _Float16* F1h = (_Float16*)take(NTF * 2);

    auto cvt = [&](const float* s, _Float16* d, size_t n) {
        int n4 = (int)(n / 4);
        k_cvt4<<<dim3((unsigned)((n4 + 255) / 256)), dim3(256), 0, stream>>>(s, d, n4);
    };

    // 1) convert everything to f16
    cvt(X, X16, NTD);
    cvt(ENC, E16, NTD);
    for (int i = 0; i < 8; ++i) cvt(Wf[i], Wh[i], (size_t)Dd * Dd);
    cvt(W1f, W1h, (size_t)FH * Dd);
    cvt(W2f, W2h, (size_t)Dd * FH);

    const dim3 wv(32);
    const dim3 gD(BT / 32, Dd / 64);     // N=512 GEMMs, 32x64 per wave
    const dim3 gF(BT / 32, FH / 64);     // N=2048 GEMM
    const dim3 gA(Tt / 16, Hh, Bb);      // attention
    const dim3 gLN(BT / 16);

    // 2) self-attn projections
    k_gemm<<<gD, wv, 0, stream>>>(X16, Wh[0], Qh,  Dd, Dd, nullptr, 0);
    k_gemm<<<gD, wv, 0, stream>>>(X16, Wh[1], Kh,  Dd, Dd, nullptr, 0);
    k_gemm<<<gD, wv, 0, stream>>>(X16, Wh[2], Vth, Dd, Dd, nullptr, 2);
    // 3) self-attention
    k_attn<<<gA, wv, 0, stream>>>(Qh, Kh, Vth, Oh, dvl, 0);
    // 4) Wo1 + residual(X) + LN1 -> Y
    k_gemm_ln<<<gLN, dim3(128), 0, stream>>>(Oh, Wh[3], Dd, nullptr, X, g1, be1, Y32, Y16);
    // 5) cross-attn projections (Q from Y, K/V from enc)
    k_gemm<<<gD, wv, 0, stream>>>(Y16, Wh[4], Qh,  Dd, Dd, nullptr, 0);
    k_gemm<<<gD, wv, 0, stream>>>(E16, Wh[5], Kh,  Dd, Dd, nullptr, 0);
    k_gemm<<<gD, wv, 0, stream>>>(E16, Wh[6], Vth, Dd, Dd, nullptr, 2);
    // 6) cross-attention
    k_attn<<<gA, wv, 0, stream>>>(Qh, Kh, Vth, Oh, evl, 1);
    // 7) Wo2 + residual(Y) + LN2 -> Z (in place over Y buffers)
    k_gemm_ln<<<gLN, dim3(128), 0, stream>>>(Oh, Wh[7], Dd, nullptr, Y32, g2, be2, Y32, Y16);
    // 8) FFN1: leaky(Z @ W1^T + b1)
    k_gemm<<<gF, wv, 0, stream>>>(Y16, W1h, F1h, FH, Dd, b1, 1);
    // 9) FFN2 + b2 + residual(Z) + LN3 -> output (fp32)
    k_gemm_ln<<<gLN, dim3(128), 0, stream>>>(F1h, W2h, FH, b2, Y32, g3, be3, OUT, nullptr);

    (void)in_sizes; (void)n_in; (void)out_size; (void)ws_size;
}
